// GCN_80788334838501
// MI455X (gfx1250) — matrix operand order repaired and checked by reading, not compiled, not run on hardware
//
#include <hip/hip_runtime.h>
#include <hip/hip_bf16.h>

// ---------------------------------------------------------------------------
// GCN (2-layer, PyG GCNConv semantics) for gfx1250 / MI455X.
//   h  = relu( Scatter(norm * (X@W1)) + b1 )
//   out =      Scatter(norm * (h@W2)) + b2
// GEMMs: v_wmma_f32_16x16x32_bf16 with hi/lo bf16 split (fp32-class accuracy,
// 3 WMMAs per K-step per column tile). Each wave owns a 16x64 output strip so
// one A-fragment conversion feeds 12 WMMAs. Message passing: float4 gather +
// f32 L2 atomics (accumulators are L2-resident: 51 MB < 192 MB L2).
// ---------------------------------------------------------------------------

typedef __attribute__((ext_vector_type(16))) __bf16 v16bf;
typedef __attribute__((ext_vector_type(8)))  float  v8f;

static constexpr int IN_CH  = 512;
static constexpr int HID_CH = 256;
static constexpr int OUT_CH = 128;

// -------------------------------- degree / norm ----------------------------

__global__ void k_init_deg(float* __restrict__ deg, int n) {
    int i = blockIdx.x * blockDim.x + threadIdx.x;
    if (i < n) deg[i] = 1.0f;                       // self-loop contribution
}

__global__ void k_deg_scatter(const int* __restrict__ dst,
                              float* __restrict__ deg, int E) {
    int e = blockIdx.x * blockDim.x + threadIdx.x;
    if (e < E) atomicAdd(&deg[dst[e]], 1.0f);
}

__global__ void k_deg_to_dinv(float* __restrict__ deg, int n) {
    int i = blockIdx.x * blockDim.x + threadIdx.x;
    if (i < n) {
        float d = deg[i];
        deg[i] = (d > 0.0f) ? rsqrtf(d) : 0.0f;     // in place: deg -> dinv
    }
}

// ------------------------ weight convert: fp32 -> bf16 hi/lo, transposed ---
// W is [K x NC] row-major fp32.  Output Bt_hi/Bt_lo are [NC x K] bf16 so a
// lane's 16-element B fragment (one column, 16 consecutive K) is contiguous.

__global__ void k_convert_w(const float* __restrict__ W,
                            __bf16* __restrict__ hi, __bf16* __restrict__ lo,
                            int K, int NC) {
    int idx = blockIdx.x * blockDim.x + threadIdx.x;
    if (idx >= K * NC) return;
    int k = idx / NC;
    int c = idx % NC;
    float  f = W[idx];
    __bf16 h = (__bf16)f;
    hi[(size_t)c * K + k] = h;
    lo[(size_t)c * K + k] = (__bf16)(f - (float)h);
}

// ------------------------------- WMMA GEMM ---------------------------------
// C[M x NC] = A[M x K] (fp32, split to bf16 hi/lo on the fly) @ Bt (bf16 hi/lo,
// transposed [NC x K]).  One wave computes a 16 x (16*CT) output strip; 8
// waves per 256-thread block.  K stepped by 32; per step: 1 A-fragment
// conversion feeds CT*3 v_wmma_f32_16x16x32_bf16.
//
// VGPR layouts used here (wave32; A and B share one k order):
//   A 16x32 bf16 : lane L holds row M=L%16; lanes 0-15 get K in
//                  {kk..kk+15} (element i = kk+i), lanes 16-31 {kk+16..kk+31}.
//   B 32x16 bf16 : lane L holds col N=L%16; element i is K = kk + 16*(L/16)+i.
//   C/D 16x16 f32: lane L, vgpr r -> M = r + 8*(L/16), N = L%16.

template <int K, int NC, int CT>
__global__ __launch_bounds__(256) void k_gemm_bf16x3(
    const float*  __restrict__ A,
    const __bf16* __restrict__ Bt_hi,
    const __bf16* __restrict__ Bt_lo,
    float*        __restrict__ C,
    int mtiles) {
    constexpr int NCG = NC / (16 * CT);             // column groups
    const int wave = blockIdx.x * 8 + (threadIdx.x >> 5);
    const int rt   = wave / NCG;
    const int cg   = wave % NCG;
    if (rt >= mtiles) return;                       // wave-uniform exit

    const int lane = threadIdx.x & 31;
    const int half = lane >> 4;                     // 0 | 1
    const int l16  = lane & 15;

    const float* arow = A + (size_t)(rt * 16 + l16) * K;
    const __bf16* bhp[CT];
    const __bf16* blp[CT];
#pragma unroll
    for (int t = 0; t < CT; ++t) {
        size_t coff = (size_t)((cg * CT + t) * 16 + l16) * K;
        bhp[t] = Bt_hi + coff;
        blp[t] = Bt_lo + coff;
    }

    v8f acc[CT] = {};
    for (int kk = 0; kk < K; kk += 32) {
        // A fragment: two contiguous 32B fp32 loads, split into bf16 hi/lo.
        v8f a0 = *(const v8f*)(arow + kk +     half * 16);
        v8f a1 = *(const v8f*)(arow + kk + 8 + half * 16);
        __builtin_prefetch(arow + kk + 32, 0, 0);   // global_prefetch_b8
        v16bf ah, al;
#pragma unroll
        for (int i = 0; i < 8; ++i) {
            float f = a0[i]; __bf16 h = (__bf16)f;
            ah[i] = h; al[i] = (__bf16)(f - (float)h);
        }
#pragma unroll
        for (int i = 0; i < 8; ++i) {
            float f = a1[i]; __bf16 h = (__bf16)f;
            ah[8 + i] = h; al[8 + i] = (__bf16)(f - (float)h);
        }
#pragma unroll
        for (int t = 0; t < CT; ++t) {
            // B fragments: one contiguous 32B bf16 load each.
            v16bf bhv = *(const v16bf*)(bhp[t] + kk + half * 16);
            v16bf blv = *(const v16bf*)(blp[t] + kk + half * 16);
            // acc += Ahi*Bhi + Ahi*Blo + Alo*Bhi   (bf16x3 fp32 emulation)
            acc[t] = __builtin_amdgcn_wmma_f32_16x16x32_bf16(
                         false, ah, false, bhv, (short)0, acc[t], false, false);
            acc[t] = __builtin_amdgcn_wmma_f32_16x16x32_bf16(
                         false, ah, false, blv, (short)0, acc[t], false, false);
            acc[t] = __builtin_amdgcn_wmma_f32_16x16x32_bf16(
                         false, al, false, bhv, (short)0, acc[t], false, false);
        }
    }

#pragma unroll
    for (int t = 0; t < CT; ++t) {
        float* crow = C + (size_t)(rt * 16 + half * 8) * NC
                        + (cg * CT + t) * 16 + l16;
#pragma unroll
        for (int r = 0; r < 8; ++r) crow[(size_t)r * NC] = acc[t][r];
    }
}

// ------------------------------ edge scatter -------------------------------
// out[dst] += in[src] * dinv[src]*dinv[dst], float4 per thread, L2 atomics.

template <int CH>
__global__ __launch_bounds__(256) void k_edge_scatter(
    const int*   __restrict__ src, const int* __restrict__ dst,
    const float* __restrict__ dinv,
    const float* __restrict__ Hin, float* __restrict__ Hout, int E) {
    constexpr int TPE = CH / 4;                     // threads per edge
    constexpr int EPB = 256 / TPE;                  // edges per block
    int e = blockIdx.x * EPB + threadIdx.x / TPE;
    if (e >= E) return;
    int c = (threadIdx.x % TPE) * 4;
    int s = src[e], d = dst[e];
    float nrm = dinv[s] * dinv[d];
    const float4 v = *(const float4*)(Hin + (size_t)s * CH + c);
    float* o = Hout + (size_t)d * CH + c;
    atomicAdd(o + 0, v.x * nrm);
    atomicAdd(o + 1, v.y * nrm);
    atomicAdd(o + 2, v.z * nrm);
    atomicAdd(o + 3, v.w * nrm);
}

// self-loop term + bias (+ optional ReLU), fused epilogue
template <int CH, bool RELU>
__global__ void k_finish(const float* __restrict__ Hpre,
                         const float* __restrict__ bias,
                         const float* __restrict__ dinv,
                         float* __restrict__ Hacc, int n) {
    int i = blockIdx.x;                             // node,  grid = n
    int c = threadIdx.x;                            // channel, block = CH
    float di = dinv[i];
    size_t off = (size_t)i * CH + c;
    float v = Hacc[off] + Hpre[off] * di * di + bias[c];
    if (RELU) v = fmaxf(v, 0.0f);
    Hacc[off] = v;
}

// ------------------------------- launcher ----------------------------------

extern "C" void kernel_launch(void* const* d_in, const int* in_sizes, int n_in,
                              void* d_out, int out_size, void* d_ws, size_t ws_size,
                              hipStream_t stream) {
    const float* x  = (const float*)d_in[0];
    const int*   ei = (const int*)  d_in[1];
    const float* W1 = (const float*)d_in[2];
    const float* b1 = (const float*)d_in[3];
    const float* W2 = (const float*)d_in[4];
    const float* b2 = (const float*)d_in[5];
    float* out = (float*)d_out;

    const int N = in_sizes[0] / IN_CH;              // 50000
    const int E = in_sizes[1] / 2;                  // 800000
    const int* src = ei;
    const int* dst = ei + E;

    // ---- workspace layout (float offsets; all regions 256B-aligned) ----
    float* ws = (float*)d_ws;
    const size_t OFF_DINV = 0;
    const size_t OFF_HW1  = 50048;                          // pre-agg layer1
    const size_t OFF_H1   = OFF_HW1 + (size_t)N * HID_CH;   // agg accumulator
    const size_t OFF_H2   = OFF_HW1;                        // alias: HW1 dead by then
    const size_t bf_base  = (OFF_H1 + (size_t)N * HID_CH) * sizeof(float);
    float* dinv = ws + OFF_DINV;
    float* HW1  = ws + OFF_HW1;
    float* H1   = ws + OFF_H1;
    float* H2   = ws + OFF_H2;
    __bf16* w1h = (__bf16*)((char*)d_ws + bf_base);
    __bf16* w1l = w1h + (size_t)IN_CH  * HID_CH;
    __bf16* w2h = w1l + (size_t)IN_CH  * HID_CH;
    __bf16* w2l = w2h + (size_t)HID_CH * OUT_CH;

    // ---- zero atomic accumulators (memset nodes are graph-capturable) ----
    hipMemsetAsync(H1,  0, (size_t)N * HID_CH * sizeof(float), stream);
    hipMemsetAsync(out, 0, (size_t)out_size * sizeof(float),  stream);

    // ---- degree -> dinv ----
    k_init_deg   <<<(N + 255) / 256, 256, 0, stream>>>(dinv, N);
    k_deg_scatter<<<(E + 255) / 256, 256, 0, stream>>>(dst, dinv, E);
    k_deg_to_dinv<<<(N + 255) / 256, 256, 0, stream>>>(dinv, N);

    // ---- weights: fp32 -> transposed bf16 hi/lo ----
    k_convert_w<<<(IN_CH * HID_CH + 255) / 256, 256, 0, stream>>>(W1, w1h, w1l, IN_CH,  HID_CH);
    k_convert_w<<<(HID_CH * OUT_CH + 255) / 256, 256, 0, stream>>>(W2, w2h, w2l, HID_CH, OUT_CH);

    const int mtiles = N / 16;                      // 3125

    // ---- layer 1: HW1 = X @ W1 ; scatter ; +self +b1, ReLU ----
    {
        int waves  = mtiles * (HID_CH / 64);        // 16x64 strip per wave
        int blocks = (waves + 7) / 8;
        k_gemm_bf16x3<IN_CH, HID_CH, 4><<<blocks, 256, 0, stream>>>(x, w1h, w1l, HW1, mtiles);
    }
    k_edge_scatter<HID_CH><<<(E + 3) / 4, 256, 0, stream>>>(src, dst, dinv, HW1, H1, E);
    k_finish<HID_CH, true><<<N, HID_CH, 0, stream>>>(HW1, b1, dinv, H1, N);

    // ---- layer 2: H2 = H1 @ W2 ; scatter ; +self +b2 ----
    {
        int waves  = mtiles * (OUT_CH / 64);
        int blocks = (waves + 7) / 8;
        k_gemm_bf16x3<HID_CH, OUT_CH, 4><<<blocks, 256, 0, stream>>>(H1, w2h, w2l, H2, mtiles);
    }
    k_edge_scatter<OUT_CH><<<(E + 7) / 8, 256, 0, stream>>>(src, dst, dinv, H2, out, E);
    k_finish<OUT_CH, false><<<N, OUT_CH, 0, stream>>>(H2, b2, dinv, out, N);
}
